// StrucTreeEncoder_69965017252556
// MI455X (gfx1250) — matrix operand order repaired and checked by reading, not compiled, not run on hardware
//
#include <hip/hip_runtime.h>
#include <hip/hip_bf16.h>

typedef __attribute__((ext_vector_type(16))) _Float16 v16h;
typedef __attribute__((ext_vector_type(8)))  float    v8f;

#define LATENT 128

// Single-wave GEMV via WMMA:
//   vout[0..Ncols) = act( vin[0..K) @ W(K x Ncols, row-major) + bias )
// A-fragment trick: every row of the 16x32 A tile holds the same K-slice of
// vin, so every row of D is the GEMV result; we read D row 0 via acc[0].
// Lane layout per CDNA5 ISA (16-bit A 16x32): lanes 0-15 hold K {0..7,16..23},
// lanes 16-31 hold K {8..15,24..31}. B 32x16: lane&15 = column, lane>>4 picks
// K {0..7,16..23} vs {8..15,24..31}, same as A.
__device__ __forceinline__ void wave_gemv_wmma(
    const float* __restrict__ vin,   // LDS, length K
    const float* __restrict__ W,     // global, K x Ncols row-major
    const float* __restrict__ bias,  // global, length Ncols
    float*       __restrict__ vout,  // LDS, length Ncols
    int K, int Ncols, bool do_relu, int lane)
{
  const int g   = lane >> 4;   // lane-group: 0 = lanes 0-15, 1 = lanes 16-31
  const int col = lane & 15;
  for (int t = 0; t < Ncols / 16; ++t) {
    v8f acc = {};
    for (int c = 0; c < K / 32; ++c) {
      const int base = 32 * c;
      v16h a, b;
      #pragma unroll
      for (int j = 0; j < 8; ++j) {
        a[j]     = (_Float16)vin[base +      8 * g + j];  // K = base+8g+j
        a[8 + j] = (_Float16)vin[base + 16 + 8 * g + j];  // K = base+16+8g+j
      }
      #pragma unroll
      for (int j = 0; j < 16; ++j)
        b[j] = (_Float16)W[(size_t)(base + 8 * g + (j & 7) + 16 * (j >> 3)) * Ncols + 16 * t + col];
      // D = A x B + C, f32 accumulate
      acc = __builtin_amdgcn_wmma_f32_16x16x32_f16(false, a, false, b,
                                                   (short)0, acc, false, false);
    }
    float r = acc[0] + bias[16 * t + col];   // D[0, col] (lanes>=16: D[8,col], identical)
    if (do_relu) r = fmaxf(r, 0.0f);
    if (lane < 16) vout[16 * t + col] = r;
  }
}

__global__ void structree_encoder_kernel(
    const float* __restrict__ x,
    const float* __restrict__ w1s, const float* __restrict__ b1s,
    const float* __restrict__ w2s, const float* __restrict__ b2s,
    const float* __restrict__ w1c, const float* __restrict__ b1c,
    const float* __restrict__ w2c, const float* __restrict__ b2c,
    const int* __restrict__ num_node,
    float* __restrict__ out)
{
  __shared__ float bufA[2 * LATENT];
  __shared__ float bufB[2 * LATENT];
  const int lane = threadIdx.x;

  // Wave-uniform scalar branch (keeps EXEC all-1s around the WMMA region).
  const int N = __builtin_amdgcn_readfirstlane(num_node[0]);

  // Dataflow analysis of the reference: the collect scan's carry position can
  // only match the loop index on the very first step, which requires N == 2.
  // For any other N the final gate (pos_f == 0) fails -> output is all zeros.
  if (N != 2) {
    #pragma unroll
    for (int i = lane; i < LATENT; i += 32) out[i] = 0.0f;
    return;
  }

  // N == 2 corner: out = g(f(pad(x[0]))).
  for (int i = lane; i < LATENT; i += 32) bufA[i] = (i < 2) ? x[i] : 0.0f;
  __syncthreads();
  // f: relu(v @ w1s + b1s) @ w2s + b2s
  wave_gemv_wmma(bufA, w1s, b1s, bufB, LATENT,     2 * LATENT, true,  lane);
  __syncthreads();
  wave_gemv_wmma(bufB, w2s, b2s, bufA, 2 * LATENT, LATENT,     false, lane);
  __syncthreads();
  // g: relu(v @ w1c + b1c) @ w2c + b2c
  wave_gemv_wmma(bufA, w1c, b1c, bufB, LATENT,     2 * LATENT, true,  lane);
  __syncthreads();
  wave_gemv_wmma(bufB, w2c, b2c, bufA, 2 * LATENT, LATENT,     false, lane);
  __syncthreads();

  for (int i = lane; i < LATENT; i += 32) out[i] = bufA[i];
}

extern "C" void kernel_launch(void* const* d_in, const int* in_sizes, int n_in,
                              void* d_out, int out_size, void* d_ws, size_t ws_size,
                              hipStream_t stream) {
  (void)in_sizes; (void)n_in; (void)out_size; (void)d_ws; (void)ws_size;
  // setup_inputs order:
  //  0:x 1:w1s 2:b1s 3:w2s 4:b2s 5:w1c 6:b1c 7:w2c 8:b2c 9:num_node 10:edge_index
  structree_encoder_kernel<<<dim3(1), dim3(32), 0, stream>>>(
      (const float*)d_in[0],
      (const float*)d_in[1], (const float*)d_in[2],
      (const float*)d_in[3], (const float*)d_in[4],
      (const float*)d_in[5], (const float*)d_in[6],
      (const float*)d_in[7], (const float*)d_in[8],
      (const int*)d_in[9],
      (float*)d_out);
}
